// EFNtoGlobal_24172075941939
// MI455X (gfx1250) — matrix operand order repaired and checked by reading, not compiled, not run on hardware
//
#include <hip/hip_runtime.h>

typedef __attribute__((ext_vector_type(16))) __bf16 v16bf;
typedef __attribute__((ext_vector_type(8)))  __bf16 v8bf;
typedef __attribute__((ext_vector_type(8)))  float  v8f;

#define DIN  64
#define DH   128   // hidden = out = 128

// native RNE float -> bf16 (gfx1250 has hardware bf16 cvt)
static __device__ __forceinline__ __bf16 f2bf(float f) {
    return (__bf16)f;
}

// ---------------------------------------------------------------------------
// Kernel 0: zero node accumulator (in d_ws) and output
// ---------------------------------------------------------------------------
extern "C" __global__ void efn_zero(float* __restrict__ hnode,
                                    float* __restrict__ out,
                                    long long nH, int nOut) {
    long long stride = (long long)gridDim.x * blockDim.x;
    long long total  = nH + (long long)nOut;
    for (long long i = (long long)blockIdx.x * blockDim.x + threadIdx.x;
         i < total; i += stride) {
        if (i < nH) hnode[i] = 0.0f;
        else        out[i - nH] = 0.0f;
    }
}

// ---------------------------------------------------------------------------
// Kernel 1: fused edge MLP + scatter-add into per-node accumulator.
// Block = 256 threads = 8 wave32; each wave processes one 16-edge tile.
// Dynamic LDS layout (bf16 elements):
//   [0 .. 16384)             W1 in B-fragment order: frag(n,kb) -> 32 lanes x 16 halves
//   [16384 .. 32768)         W2 same
//   [32768 .. 32768+8*2048)  per-wave 16x128 hidden staging tile
// ---------------------------------------------------------------------------
extern "C" __global__ __launch_bounds__(256)
void efn_edge_mlp(const float* __restrict__ x,
                  const int*   __restrict__ src,
                  const int*   __restrict__ dst,
                  const float* __restrict__ W1, const float* __restrict__ b1,
                  const float* __restrict__ W2, const float* __restrict__ b2,
                  float* __restrict__ hnode, int nEdges) {
    extern __shared__ __bf16 smem[];
    __bf16* sW1 = smem;
    __bf16* sW2 = smem + 16384;

    const int t = threadIdx.x;

    // ---- cooperative load + swizzle of W1/W2 into LDS B-fragment layout ----
    // B frag(n,kb) element (lane,h):  W[kb*32 + (h>>3)*16 + (lane>>4)*8 + (h&7)][n*16 + (lane&15)]
    #pragma unroll 4
    for (int i = 0; i < 64; ++i) {
        int idx  = (i << 8) + t;          // coalesced over 128x128
        int k    = idx >> 7;              // row of W (K index)
        int col  = idx & 127;             // column of W (N index)
        int n    = col >> 4;
        int kb   = k >> 5;
        int lane = (((k >> 3) & 1) << 4) | (col & 15);
        int h    = (k & 7) | (((k >> 4) & 1) << 3);
        int off  = (((n << 2) + kb) << 5) * 16 + lane * 16 + h;
        sW1[off] = f2bf(W1[idx]);
        sW2[off] = f2bf(W2[idx]);
    }
    __syncthreads();

    const int wave = t >> 5;
    const int lane = t & 31;
    const int m    = lane & 15;       // edge row within tile / output column
    const int hi   = lane >> 4;       // half-wave select
    const int klo  = hi << 3;         // 0 or 8
    __bf16* sHid = smem + 32768 + wave * (16 * DH);

    const long long e0 = ((long long)blockIdx.x * 8 + wave) * 16;
    if (e0 >= nEdges) return;         // wave-uniform exit

    int eg = (int)e0 + m;
    if (eg >= nEdges) eg = nEdges - 1;   // clamp gathers on (theoretical) tail
    const int vi = dst[eg];
    const int vj = src[eg];
    const float* xi = x + (long long)vi * DIN;
    const float* xj = x + (long long)vj * DIN;

    // gather the 32 floats of x_i / x_j this lane contributes to A fragments
    float fi[32], fj[32];
    #pragma unroll
    for (int q = 0; q < 4; ++q) {
        int base = q * 16 + klo;
        *(float4*)&fi[q * 8]     = *(const float4*)(xi + base);
        *(float4*)&fi[q * 8 + 4] = *(const float4*)(xi + base + 4);
        *(float4*)&fj[q * 8]     = *(const float4*)(xj + base);
        *(float4*)&fj[q * 8 + 4] = *(const float4*)(xj + base + 4);
    }

    // A fragments for GEMM1: msg = [x_i , x_j - x_i], 16x128 bf16 (4 K-blocks)
    v16bf A[4];
    #pragma unroll
    for (int j = 0; j < 8; ++j) {
        A[0][j]     = f2bf(fi[j]);
        A[0][j + 8] = f2bf(fi[8 + j]);
        A[1][j]     = f2bf(fi[16 + j]);
        A[1][j + 8] = f2bf(fi[24 + j]);
        A[2][j]     = f2bf(fj[j]      - fi[j]);
        A[2][j + 8] = f2bf(fj[8 + j]  - fi[8 + j]);
        A[3][j]     = f2bf(fj[16 + j] - fi[16 + j]);
        A[3][j + 8] = f2bf(fj[24 + j] - fi[24 + j]);
    }

    // ---- GEMM1: hidden = relu(msg @ W1 + b1), stored bf16 to LDS ----------
    // First WMMA of each chain uses inline-0 C; bias folded into the epilogue.
    #pragma unroll
    for (int n = 0; n < 8; ++n) {
        v8f c = {};                       // inline 0 for first WMMA
        #pragma unroll
        for (int kb = 0; kb < 4; ++kb) {
            const v16bf B = *(const v16bf*)(sW1 + (((n << 2) + kb) * 32 + lane) * 16);
            c = __builtin_amdgcn_wmma_f32_16x16x32_bf16(
                    false, A[kb], false, B, (short)0, c, false, false);
        }
        const float bias = b1[n * 16 + m];
        #pragma unroll
        for (int r = 0; r < 8; ++r) {
            float v = c[r] + bias;
            v = v > 0.0f ? v : 0.0f;                        // ReLU
            sHid[(hi * 8 + r) * DH + n * 16 + m] = f2bf(v); // C-layout -> row major
        }
    }

    // wave-local LDS RAW: make the transposed reload explicit
    asm volatile("s_wait_dscnt 0" ::: "memory");

    // A fragments for GEMM2 from the staged hidden tile (layout transpose)
    v16bf H[4];
    #pragma unroll
    for (int kb = 0; kb < 4; ++kb) {
        const __bf16* hp = sHid + m * DH + kb * 32 + klo;
        v8bf lo = *(const v8bf*)hp;
        v8bf hh = *(const v8bf*)(hp + 16);
        #pragma unroll
        for (int j = 0; j < 8; ++j) { H[kb][j] = lo[j]; H[kb][j + 8] = hh[j]; }
    }

    // destination node of each of my 8 output rows
    int dste[8];
    #pragma unroll
    for (int r = 0; r < 8; ++r) {
        int er = (int)e0 + hi * 8 + r;
        dste[r] = dst[er < nEdges ? er : (nEdges - 1)];
    }

    // ---- GEMM2: msg_out = hidden @ W2 + b2, scatter-add by dst ------------
    #pragma unroll
    for (int n = 0; n < 8; ++n) {
        v8f c = {};                       // inline 0 for first WMMA
        #pragma unroll
        for (int kb = 0; kb < 4; ++kb) {
            const v16bf B = *(const v16bf*)(sW2 + (((n << 2) + kb) * 32 + lane) * 16);
            c = __builtin_amdgcn_wmma_f32_16x16x32_bf16(
                    false, H[kb], false, B, (short)0, c, false, false);
        }
        const float bias = b2[n * 16 + m];
        #pragma unroll
        for (int r = 0; r < 8; ++r) {
            int er = (int)e0 + hi * 8 + r;
            if (er < nEdges)
                atomicAdd(&hnode[(long long)dste[r] * DH + n * 16 + m], c[r] + bias);
        }
    }
}

// ---------------------------------------------------------------------------
// Kernel 2: out[batch[v]] += p[v,0] * hnode[v]   (batch is sorted ascending)
// One thread per output column; register accumulation, atomic flush only at
// graph boundaries.
// ---------------------------------------------------------------------------
extern "C" __global__ void efn_scale_segment(const float* __restrict__ p,
                                             const int*   __restrict__ batch,
                                             const float* __restrict__ hnode,
                                             float* __restrict__ out,
                                             int nNodes, int chunk) {
    const int col = threadIdx.x;                 // 0..127
    int v0 = blockIdx.x * chunk;
    if (v0 >= nNodes) return;
    int v1 = v0 + chunk; if (v1 > nNodes) v1 = nNodes;

    int g = batch[v0];
    float acc = 0.0f;
    for (int v = v0; v < v1; ++v) {
        int b = batch[v];
        if (b != g) {
            atomicAdd(&out[g * DH + col], acc);
            acc = 0.0f;
            g = b;
        }
        acc += p[v * 4] * hnode[(long long)v * DH + col];
    }
    atomicAdd(&out[g * DH + col], acc);
}

// ---------------------------------------------------------------------------
extern "C" void kernel_launch(void* const* d_in, const int* in_sizes, int n_in,
                              void* d_out, int out_size, void* d_ws, size_t ws_size,
                              hipStream_t stream) {
    const float* x     = (const float*)d_in[0];
    const float* p     = (const float*)d_in[1];
    const int*   ei    = (const int*)  d_in[2];   // (2, E): row0=src, row1=dst
    const int*   batch = (const int*)  d_in[3];
    const float* W1    = (const float*)d_in[4];
    const float* b1    = (const float*)d_in[5];
    const float* W2    = (const float*)d_in[6];
    const float* b2    = (const float*)d_in[7];
    float* out = (float*)d_out;

    const int nNodes = in_sizes[0] / DIN;
    const int nEdges = in_sizes[2] / 2;
    const int* src = ei;
    const int* dst = ei + nEdges;

    float* hnode = (float*)d_ws;                        // nNodes x 128 f32
    const long long nH = (long long)nNodes * DH;

    efn_zero<<<2048, 256, 0, stream>>>(hnode, out, nH, out_size);

    const int tilesPerBlock = 8;                        // 8 waves/block
    const int nTiles  = (nEdges + 15) / 16;
    const int nBlocks = (nTiles + tilesPerBlock - 1) / tilesPerBlock;
    const size_t shmem = (size_t)(16384 + 16384 + 8 * 16 * DH) * sizeof(__bf16); // 96 KB
    efn_edge_mlp<<<nBlocks, 256, shmem, stream>>>(x, src, dst, W1, b1, W2, b2,
                                                  hnode, nEdges);

    const int chunk = 128;
    const int nb2 = (nNodes + chunk - 1) / chunk;
    efn_scale_segment<<<nb2, DH, 0, stream>>>(p, batch, hnode, out, nNodes, chunk);
}
